// ReadUnit_1357209665576
// MI455X (gfx1250) — compile-verified, matrix operand order checked by reading, not run on hardware
//
#include <hip/hip_runtime.h>
#include <hip/hip_bf16.h>

typedef __bf16 v16bf __attribute__((ext_vector_type(16)));
typedef __bf16 v8bf  __attribute__((ext_vector_type(8)));
typedef __bf16 v4bf  __attribute__((ext_vector_type(4)));
typedef float  v8f   __attribute__((ext_vector_type(8)));

#define DDIM   128
#define BATCH  16
#define KTOT   8192
#define TK     128
#define NBLK   (KTOT / TK)   // 64

// ---- workspace layout (bytes) ----
#define IM_OFF    0u            // 16*128 f32      = 8192 B
#define WIK_OFF   8192u         // 128*128 bf16    = 32768 B
#define WID_OFF   40960u        // 256*128 bf16    = 65536 B
#define WRA_OFF   106496u       // 128*128 bf16    = 32768 B
#define PMAX_OFF  139264u       // 16*64*128 f32   = 524288 B
#define PSUM_OFF  663552u
#define PR_OFF    1187840u
// total ~1.72 MB

// ---------------- fragment helpers ----------------
// A-fragment (bf16 16x32, M x K) from row-major bf16 LDS buffer [TK][128]:
// lane 0-15: M=lane, K = ks*32 + {0..7, 16..23}; lanes 16-31: K shifted by +8.
__device__ inline v16bf loadA(const __bf16* buf, int m, int kstep, int hi) {
    const __bf16* row = buf + m * DDIM + kstep * 32 + hi * 8;
    v8bf c0 = *(const v8bf*)(row);
    v8bf c1 = *(const v8bf*)(row + 16);
    return __builtin_shufflevector(c0, c1, 0,1,2,3,4,5,6,7,8,9,10,11,12,13,14,15);
}

// B-fragment: prepacked contiguous 16 bf16 per lane per (ntile,kstep)
__device__ inline v16bf loadB(const __bf16* frag, int ntile, int ksteps, int kstep, int lane) {
    return *(const v16bf*)(frag + ((((size_t)ntile * ksteps + kstep) * 32 + lane) << 4));
}

// pack weight W (Krows x 128, row-major f32) into B-fragment order bf16
__device__ inline void packFrag(const float* __restrict__ W, __bf16* __restrict__ dst,
                                int ksteps, int tid) {
    int total = 8 * ksteps * 32 * 16;
    for (int f = tid; f < total; f += 256) {
        int i    = f & 15;
        int lane = (f >> 4) & 31;
        int rest = f >> 9;
        int ks   = rest % ksteps;
        int nt   = rest / ksteps;
        int hi   = (lane >> 4) & 1;
        int kl   = ((i < 8) ? i : (i + 8)) + hi * 8;
        int k    = ks * 32 + kl;
        int n    = nt * 16 + (lane & 15);
        dst[f] = (__bf16)W[k * DDIM + n];
    }
}

// ---------------- prep kernel ----------------
__global__ void readunit_prep(const float* __restrict__ m_prev,
                              const float* __restrict__ im_kernel,
                              const float* __restrict__ im_bias,
                              const float* __restrict__ ik_kernel,
                              const float* __restrict__ id_kernel,
                              const float* __restrict__ ra_kernel,
                              float* __restrict__ wsIm,
                              __bf16* __restrict__ wik,
                              __bf16* __restrict__ wid,
                              __bf16* __restrict__ wra) {
    int tid = threadIdx.x;
    // Im = m_prev @ im_kernel + im_bias  (16x128, tiny)
    for (int idx = tid; idx < BATCH * DDIM; idx += 256) {
        int b = idx >> 7, e = idx & 127;
        float s = im_bias[e];
        for (int k = 0; k < DDIM; ++k)
            s += m_prev[b * DDIM + k] * im_kernel[k * DDIM + e];
        wsIm[idx] = s;
    }
    packFrag(ik_kernel, wik, 4, tid);
    packFrag(id_kernel, wid, 8, tid);   // 256 rows: ks 0..3 = I part, 4..7 = knowledge part
    packFrag(ra_kernel, wra, 4, tid);
}

// ---------------- main fused kernel ----------------
__global__ __launch_bounds__(256)
void readunit_main(const float* __restrict__ c_cur,
                   const float* __restrict__ knowledge,
                   const float* __restrict__ ik_bias,
                   const float* __restrict__ id_bias,
                   const float* __restrict__ wsIm,
                   const __bf16* __restrict__ wik,
                   const __bf16* __restrict__ wid,
                   const __bf16* __restrict__ wra,
                   float* __restrict__ pmax,
                   float* __restrict__ psum,
                   float* __restrict__ pr) {
    __shared__ __bf16 knS[TK * DDIM];   // 32 KB knowledge tile (bf16)
    __shared__ __bf16 tmpS[TK * DDIM];  // 32 KB: I, then cI (bf16)
    __shared__ float  raS[TK * DDIM];   // 64 KB: ra logits (f32)
    __shared__ float  redS[3 * 256];    //  3 KB: softmax partial merge

    const int kblk = blockIdx.x;
    const int b    = blockIdx.y;
    const int tid  = threadIdx.x;
    const int lane = tid & 31;
    const int wave = tid >> 5;
    const int hi   = (lane >> 4) & 1;
    const int mA   = wave * 16 + (lane & 15);   // A-fragment row for this wave

    // ---- stage knowledge tile: global f32 -> LDS bf16 ----
    const float4* knG = (const float4*)(knowledge + ((size_t)b * KTOT + (size_t)kblk * TK) * DDIM);
    #pragma unroll
    for (int j = 0; j < 16; ++j) {
        int e = tid + j * 256;          // float4 index, 4096 total
        float4 vv = knG[e];
        v4bf h;
        h[0] = (__bf16)vv.x; h[1] = (__bf16)vv.y;
        h[2] = (__bf16)vv.z; h[3] = (__bf16)vv.w;
        *(v4bf*)(knS + e * 4) = h;
    }
    __syncthreads();

    // knowledge A-fragments: hoisted once, reused by stages 1 and 2
    v16bf aKn[4];
    #pragma unroll
    for (int ks = 0; ks < 4; ++ks) aKn[ks] = loadA(knS, mA, ks, hi);

    // ---- stage 1: Ik = kn @ Wik ; I = Im * (Ik + ik_bias) -> tmpS (bf16) ----
    #pragma unroll
    for (int nt = 0; nt < 8; ++nt) {
        int n = nt * 16 + (lane & 15);
        v8f acc = {};
        #pragma unroll
        for (int ks = 0; ks < 4; ++ks) {
            v16bf bb = loadB(wik, nt, 4, ks, lane);
            acc = __builtin_amdgcn_wmma_f32_16x16x32_bf16(false, aKn[ks], false, bb,
                                                          (short)0, acc, false, false);
        }
        float f  = wsIm[b * DDIM + n];
        float bi = ik_bias[n];
        #pragma unroll
        for (int v = 0; v < 8; ++v) {
            int mv = wave * 16 + v + hi * 8;   // D-matrix row
            tmpS[mv * DDIM + n] = (__bf16)(f * (acc[v] + bi));
        }
    }
    __syncthreads();

    // ---- stage 2: Id = I @ Wid_top + kn @ Wid_bot ; cI = c_cur * (Id + id_bias) ----
    v16bf aI[4];
    #pragma unroll
    for (int ks = 0; ks < 4; ++ks) aI[ks] = loadA(tmpS, mA, ks, hi);

    float creg[64];
    #pragma unroll
    for (int nt = 0; nt < 8; ++nt) {
        int n = nt * 16 + (lane & 15);
        v8f acc = {};
        #pragma unroll
        for (int ks = 0; ks < 4; ++ks) {
            v16bf bb = loadB(wid, nt, 8, ks, lane);
            acc = __builtin_amdgcn_wmma_f32_16x16x32_bf16(false, aI[ks], false, bb,
                                                          (short)0, acc, false, false);
        }
        #pragma unroll
        for (int ks = 0; ks < 4; ++ks) {
            v16bf bb = loadB(wid, nt, 8, ks + 4, lane);
            acc = __builtin_amdgcn_wmma_f32_16x16x32_bf16(false, aKn[ks], false, bb,
                                                          (short)0, acc, false, false);
        }
        float cc = c_cur[b * DDIM + n];
        float bi = id_bias[n];
        #pragma unroll
        for (int v = 0; v < 8; ++v)
            creg[nt * 8 + v] = cc * (acc[v] + bi);
    }
    __syncthreads();   // everyone done reading I from tmpS
    #pragma unroll
    for (int nt = 0; nt < 8; ++nt) {
        int n = nt * 16 + (lane & 15);
        #pragma unroll
        for (int v = 0; v < 8; ++v) {
            int mv = wave * 16 + v + hi * 8;
            tmpS[mv * DDIM + n] = (__bf16)creg[nt * 8 + v];
        }
    }
    __syncthreads();

    // ---- stage 3: ra = cI @ Wra  (ra_bias is constant along softmax axis -> cancels) ----
    v16bf aCI[4];
    #pragma unroll
    for (int ks = 0; ks < 4; ++ks) aCI[ks] = loadA(tmpS, mA, ks, hi);

    #pragma unroll
    for (int nt = 0; nt < 8; ++nt) {
        int n = nt * 16 + (lane & 15);
        v8f acc = {};
        #pragma unroll
        for (int ks = 0; ks < 4; ++ks) {
            v16bf bb = loadB(wra, nt, 4, ks, lane);
            acc = __builtin_amdgcn_wmma_f32_16x16x32_bf16(false, aCI[ks], false, bb,
                                                          (short)0, acc, false, false);
        }
        #pragma unroll
        for (int v = 0; v < 8; ++v) {
            int mv = wave * 16 + v + hi * 8;
            raS[mv * DDIM + n] = acc[v];
        }
    }
    __syncthreads();

    // ---- per-block online-softmax partials over this 128-row tile ----
    // 2 threads per column: thread (h, n) handles rows [h*64, h*64+64)
    {
        int n = tid & 127;
        int h = tid >> 7;
        float M = -3.0e38f, S = 0.f, R = 0.f;
        int m0 = h * 64;
        for (int m = m0; m < m0 + 64; ++m) {
            float x  = raS[m * DDIM + n];
            float kv = (float)knS[m * DDIM + n];
            float Mn = fmaxf(M, x);
            float sc = __expf(M - Mn);   // first iter underflows to 0 (finite sentinel)
            float e  = __expf(x - Mn);
            S = S * sc + e;
            R = R * sc + e * kv;
            M = Mn;
        }
        redS[tid]       = M;
        redS[256 + tid] = S;
        redS[512 + tid] = R;
    }
    __syncthreads();
    if (tid < 128) {
        int n = tid;
        float M1 = redS[n],           S1 = redS[256 + n],       R1 = redS[512 + n];
        float M2 = redS[n + 128],     S2 = redS[384 + n],       R2 = redS[640 + n];
        float Mn = fmaxf(M1, M2);
        float s1 = __expf(M1 - Mn);
        float s2 = __expf(M2 - Mn);
        float S  = S1 * s1 + S2 * s2;
        float R  = R1 * s1 + R2 * s2;
        size_t o = ((size_t)b * NBLK + kblk) * DDIM + n;
        pmax[o] = Mn; psum[o] = S; pr[o] = R;
    }
}

// ---------------- combine kernel ----------------
__global__ void readunit_combine(const float* __restrict__ pmax,
                                 const float* __restrict__ psum,
                                 const float* __restrict__ pr,
                                 float* __restrict__ out) {
    int b = blockIdx.x;
    int n = threadIdx.x;
    float M = -3.0e38f;
    for (int i = 0; i < NBLK; ++i)
        M = fmaxf(M, pmax[((size_t)b * NBLK + i) * DDIM + n]);
    float S = 0.f, R = 0.f;
    for (int i = 0; i < NBLK; ++i) {
        size_t o = ((size_t)b * NBLK + i) * DDIM + n;
        float sc = __expf(pmax[o] - M);
        S += psum[o] * sc;
        R += pr[o] * sc;
    }
    out[b * DDIM + n] = R / S;
}

extern "C" void kernel_launch(void* const* d_in, const int* in_sizes, int n_in,
                              void* d_out, int out_size, void* d_ws, size_t ws_size,
                              hipStream_t stream) {
    const float* c_cur     = (const float*)d_in[0];
    const float* m_prev    = (const float*)d_in[1];
    const float* knowledge = (const float*)d_in[2];
    const float* im_kernel = (const float*)d_in[3];
    const float* ik_kernel = (const float*)d_in[4];
    const float* id_kernel = (const float*)d_in[5];
    const float* ra_kernel = (const float*)d_in[6];
    const float* im_bias   = (const float*)d_in[7];
    const float* ik_bias   = (const float*)d_in[8];
    const float* id_bias   = (const float*)d_in[9];
    // ra_bias (d_in[10]) cancels in the softmax over axis k

    char* ws = (char*)d_ws;
    float*  wsIm = (float*)(ws + IM_OFF);
    __bf16* wik  = (__bf16*)(ws + WIK_OFF);
    __bf16* wid  = (__bf16*)(ws + WID_OFF);
    __bf16* wra  = (__bf16*)(ws + WRA_OFF);
    float*  pmax = (float*)(ws + PMAX_OFF);
    float*  psum = (float*)(ws + PSUM_OFF);
    float*  pr   = (float*)(ws + PR_OFF);

    readunit_prep<<<1, 256, 0, stream>>>(m_prev, im_kernel, im_bias,
                                         ik_kernel, id_kernel, ra_kernel,
                                         wsIm, wik, wid, wra);

    dim3 grid(NBLK, BATCH);
    readunit_main<<<grid, 256, 0, stream>>>(c_cur, knowledge, ik_bias, id_bias,
                                            wsIm, wik, wid, wra, pmax, psum, pr);

    readunit_combine<<<BATCH, DDIM, 0, stream>>>(pmax, psum, pr, (float*)d_out);
}
